// LinearAttention_6631429505622
// MI455X (gfx1250) — matrix operand order repaired and checked by reading, not compiled, not run on hardware
//
#include <hip/hip_runtime.h>

// ---------------------------------------------------------------------------
// LinearAttention for MI455X (gfx1250, wave32, WMMA bf16 16x16x32 + TDM)
// ---------------------------------------------------------------------------
#define B_    8
#define C_    512
#define NPOS  4096      // 64*64 spatial positions
#define HID   1536      // 3*512
#define DH    64
#define SQRT_C 22.62741699796952f   // sqrt(512)

typedef __bf16 bfx8  __attribute__((ext_vector_type(8)));
typedef __bf16 bfx16 __attribute__((ext_vector_type(16)));
typedef float  fx8   __attribute__((ext_vector_type(8)));
typedef unsigned int u32x4 __attribute__((ext_vector_type(4)));
typedef int i32x4 __attribute__((ext_vector_type(4)));
typedef int i32x8 __attribute__((ext_vector_type(8)));

__device__ __forceinline__ float bf2f(__bf16 v) {
  unsigned short u = __builtin_bit_cast(unsigned short, v);
  unsigned int x = ((unsigned int)u) << 16;
  return __builtin_bit_cast(float, x);
}
__device__ __forceinline__ __bf16 f2bf(float f) {
  unsigned int x = __builtin_bit_cast(unsigned int, f);
  x = x + 0x7fffu + ((x >> 16) & 1u);            // round-to-nearest-even
  unsigned short u = (unsigned short)(x >> 16);
  return __builtin_bit_cast(__bf16, u);
}
// Two contiguous 16B loads -> one 16x32 (A) or 32x16 (B) bf16 fragment slice.
__device__ __forceinline__ bfx16 load_frag(const __bf16* p0, const __bf16* p1) {
  bfx8 lo = *(const bfx8*)p0;
  bfx8 hi = *(const bfx8*)p1;
  bfx16 r;
#pragma unroll
  for (int i = 0; i < 8; ++i) { r[i] = lo[i]; r[i + 8] = hi[i]; }
  return r;
}
__device__ __forceinline__ fx8 wmma_bf16(bfx16 a, bfx16 b, fx8 c) {
  return __builtin_amdgcn_wmma_f32_16x16x32_bf16(false, a, false, b,
                                                 (short)0, c, false, false);
}
__device__ __forceinline__ fx8 fx8_zero() {
  fx8 z = {0.f, 0.f, 0.f, 0.f, 0.f, 0.f, 0.f, 0.f};
  return z;
}

// ---------------------------------------------------------------------------
// TDM: 2-D tiled DMA global -> LDS with LDS row padding.
//   tile: tile_rows rows x row_dwords dwords, source row pitch stride_dwords,
//   LDS pitch = row_dwords + 4 dwords (pad 4 dwords every 16 dwords).
// D# per CDNA5 ISA: group0 = {ctl, lds_addr, gaddr_lo,
// gaddr_hi|type=2}; group1 packs data_size/pad/tensor/tile dims/strides.
// ---------------------------------------------------------------------------
__device__ __forceinline__ void tdm_load_2d(unsigned lds_addr, const void* gptr,
                                            unsigned tile_rows, unsigned row_dwords,
                                            unsigned stride_dwords) {
  unsigned long long ga = (unsigned long long)(uintptr_t)gptr;
  u32x4 g0;
  g0[0] = 1u;                                           // count=1, user mode
  g0[1] = lds_addr;                                     // lds_addr [63:32]
  g0[2] = (unsigned)ga;                                 // global_addr [95:64]
  g0[3] = (unsigned)((ga >> 32) & 0x01FFFFFFu) | (2u << 30);  // addr hi | type=2
  unsigned td0 = row_dwords;    // tensor_dim0 == tile width (no OOB)
  unsigned td1 = tile_rows;     // tensor_dim1 == tile rows
  i32x8 g1;
  // data_size=2 (4B), pad_enable, pad_interval=3 (16 dw), pad_amount=3 (4 dw)
  g1[0] = (int)((2u << 16) | (1u << 20) | (3u << 22) | (3u << 25));
  g1[1] = (int)((td0 & 0xFFFFu) << 16);                 // tensor_dim0[15:0] @48
  g1[2] = (int)((td0 >> 16) | ((td1 & 0xFFFFu) << 16)); // td0 hi | td1 lo
  g1[3] = (int)((td1 >> 16) | (row_dwords << 16));      // td1 hi | tile_dim0
  g1[4] = (int)tile_rows;                               // tile_dim1 (tile_dim2=0)
  g1[5] = (int)stride_dwords;                           // tensor_dim0_stride lo
  g1[6] = 0;                                            // stride0 hi | stride1 lo
  g1[7] = 0;
  i32x4 z4 = {0, 0, 0, 0};
#if __clang_major__ >= 23
  i32x8 z8 = {0, 0, 0, 0, 0, 0, 0, 0};
  __builtin_amdgcn_tensor_load_to_lds(g0, g1, z4, z4, z8, 0);
#else
  __builtin_amdgcn_tensor_load_to_lds(g0, g1, z4, z4, 0);
#endif
}

// ---------------------------------------------------------------------------
// K0: cast weights to bf16; fold g1*sqrt(C) into w_qkv columns.
// ---------------------------------------------------------------------------
__global__ void k_cvt_weights(const float* __restrict__ wqkv,
                              const float* __restrict__ g1,
                              const float* __restrict__ wout,
                              __bf16* __restrict__ wq, __bf16* __restrict__ wo) {
  int idx = blockIdx.x * 256 + threadIdx.x;
  if (idx < HID * C_) {
    int c = idx & (C_ - 1);
    wq[idx] = f2bf(wqkv[idx] * g1[c] * SQRT_C);
  }
  int j = idx - HID * C_;
  if (j >= 0 && j < C_ * C_) wo[j] = f2bf(wout[j]);
}

// ---------------------------------------------------------------------------
// K1: RMS-norm over channels; write bf16 TRANSPOSED xnT[b][p][c].
// ---------------------------------------------------------------------------
__global__ void k_norm_x(const float* __restrict__ x, __bf16* __restrict__ xnT) {
  int b = blockIdx.y;
  int p = blockIdx.x * 256 + threadIdx.x;
  const float* xp = x + (size_t)b * C_ * NPOS + p;
  float ss = 0.f;
  for (int c = 0; c < C_; ++c) { float v = xp[(size_t)c * NPOS]; ss += v * v; }
  float inv = rsqrtf(ss + 1e-12f);
  __bf16* o = xnT + ((size_t)b * NPOS + p) * C_;
  for (int c = 0; c < C_; ++c) o[c] = f2bf(xp[(size_t)c * NPOS] * inv);
}

// ---------------------------------------------------------------------------
// K2/K7: GEMM  C[o,p] = sum_c A[o,c] * BT[p,c]   (K = 512, bf16 in, f32 acc)
// Block: 256 thr = 8 waves; tile 128(M) x 256(N); wave tile 64x64 = 16 WMMA.
// Operand panels staged through LDS by the Tensor Data Mover, double-buffered:
//   wave 0 issues tensor_load_to_lds for step s+1, waits TENSORcnt<=2 (in-order
//   per wave => step s landed), barrier, all 8 waves read fragments from LDS.
// LDS pitch 80B (64B row + 16B TDM pad) => 16 fragment lanes hit 16 banks.
// MODE 0: bf16 out split into q/k/v slabs [3][B][512][NPOS];  MODE 1: f32 out.
// ---------------------------------------------------------------------------
template <int MODE>
__global__ __launch_bounds__(256)
void k_gemm(const __bf16* __restrict__ A, const __bf16* __restrict__ BTbase,
            void* __restrict__ Cbase) {
  constexpr int K = C_;
  constexpr int NSTEP = K / 32;
  constexpr int PITCH = 80;                    // bytes per padded LDS row
  __shared__ __align__(16) unsigned char ldsA[2][128 * PITCH];
  __shared__ __align__(16) unsigned char ldsB[2][256 * PITCH];

  int b  = blockIdx.z;
  int m0 = blockIdx.y * 128;
  int n0 = blockIdx.x * 256;
  const __bf16* BT = BTbase + (size_t)b * NPOS * K;
  int w  = threadIdx.x >> 5;
  int L  = threadIdx.x & 31;
  int lm = L & 15;
  int lh = L >> 4;
  int mw = (w & 1) * 64;        // wave tile offset inside 128x256 block tile
  int nw = (w >> 1) * 64;

  if (w == 0) {   // prologue: panels for k-step 0
    tdm_load_2d((unsigned)(uintptr_t)&ldsA[0][0], A + (size_t)m0 * K, 128, 16, 256);
    tdm_load_2d((unsigned)(uintptr_t)&ldsB[0][0], BT + (size_t)n0 * K, 256, 16, 256);
  }

  fx8 acc[4][4];
#pragma unroll
  for (int mi = 0; mi < 4; ++mi)
#pragma unroll
    for (int ni = 0; ni < 4; ++ni) acc[mi][ni] = fx8_zero();

  for (int s = 0; s < NSTEP; ++s) {
    int cur = s & 1, nxt = cur ^ 1;
    if (w == 0) {
      if (s + 1 < NSTEP) {
        int k1 = (s + 1) * 32;
        tdm_load_2d((unsigned)(uintptr_t)&ldsA[nxt][0], A + (size_t)m0 * K + k1, 128, 16, 256);
        tdm_load_2d((unsigned)(uintptr_t)&ldsB[nxt][0], BT + (size_t)n0 * K + k1, 256, 16, 256);
        __builtin_amdgcn_s_wait_tensorcnt((short)2);   // step s panels landed
      } else {
        __builtin_amdgcn_s_wait_tensorcnt((short)0);
      }
    }
    __syncthreads();            // panels [cur] visible to all waves

    bfx16 af[4], bfr[4];
#pragma unroll
    for (int mi = 0; mi < 4; ++mi) {
      const __bf16* pa = (const __bf16*)&ldsA[cur][(mw + mi * 16 + lm) * PITCH + lh * 16];
      af[mi] = load_frag(pa, pa + 16);
    }
#pragma unroll
    for (int ni = 0; ni < 4; ++ni) {
      const __bf16* pb = (const __bf16*)&ldsB[cur][(nw + ni * 16 + lm) * PITCH + lh * 16];
      bfr[ni] = load_frag(pb, pb + 16);
    }
#pragma unroll
    for (int mi = 0; mi < 4; ++mi)
#pragma unroll
      for (int ni = 0; ni < 4; ++ni)
        acc[mi][ni] = wmma_bf16(af[mi], bfr[ni], acc[mi][ni]);

    __syncthreads();            // done reading [cur]; wave 0 may refill it
  }

#pragma unroll
  for (int mi = 0; mi < 4; ++mi)
#pragma unroll
    for (int ni = 0; ni < 4; ++ni)
#pragma unroll
      for (int i = 0; i < 8; ++i) {
        int om = m0 + mw + mi * 16 + lh * 8 + i;
        int on = n0 + nw + ni * 16 + lm;
        if (MODE == 0) {
          size_t off = ((size_t)(om >> 9) * (B_ * C_) +
                        (size_t)b * C_ + (om & 511)) * NPOS + on;
          ((__bf16*)Cbase)[off] = f2bf(acc[mi][ni][i]);
        } else {
          ((float*)Cbase)[((size_t)b * C_ + om) * NPOS + on] = acc[mi][ni][i];
        }
      }
}

// ---------------------------------------------------------------------------
// K3: softmax over d (64 head channels) of q, scale by 1/8, write qT[b][p][c].
// ---------------------------------------------------------------------------
__global__ void k_softmax_q(const __bf16* __restrict__ qkv, __bf16* __restrict__ qT) {
  int b = blockIdx.z, h = blockIdx.y;
  int p = blockIdx.x * 256 + threadIdx.x;
  const __bf16* src = qkv + ((size_t)b * C_ + h * DH) * NPOS + p;  // q slab
  float v[DH];
  float m = -1e30f;
#pragma unroll
  for (int c = 0; c < DH; ++c) { v[c] = bf2f(src[(size_t)c * NPOS]); m = fmaxf(m, v[c]); }
  float s = 0.f;
#pragma unroll
  for (int c = 0; c < DH; ++c) { v[c] = __expf(v[c] - m); s += v[c]; }
  float r = 0.125f / s;   // 1/SCALE = 1/sqrt(64)
  __bf16* dst = qT + ((size_t)b * NPOS + p) * C_ + h * DH;
#pragma unroll
  for (int c = 0; c < DH; ++c) dst[c] = f2bf(v[c] * r);
}

// ---------------------------------------------------------------------------
// K4: softmax over n (4096) of each k row, in place.
// ---------------------------------------------------------------------------
__global__ void k_softmax_k(__bf16* __restrict__ qkv) {
  int b = blockIdx.y, c = blockIdx.x;
  __bf16* row = qkv + ((size_t)(B_ * C_) + (size_t)b * C_ + c) * NPOS;  // k slab
  int t = threadIdx.x;
  __shared__ float red[256];
  float vals[16];
  float m = -1e30f;
#pragma unroll
  for (int i = 0; i < 16; ++i) { vals[i] = bf2f(row[t + i * 256]); m = fmaxf(m, vals[i]); }
  red[t] = m; __syncthreads();
  for (int s = 128; s > 0; s >>= 1) { if (t < s) red[t] = fmaxf(red[t], red[t + s]); __syncthreads(); }
  m = red[0]; __syncthreads();
  float sum = 0.f;
#pragma unroll
  for (int i = 0; i < 16; ++i) { vals[i] = __expf(vals[i] - m); sum += vals[i]; }
  red[t] = sum; __syncthreads();
  for (int s = 128; s > 0; s >>= 1) { if (t < s) red[t] += red[t + s]; __syncthreads(); }
  float r = 1.f / red[0];
#pragma unroll
  for (int i = 0; i < 16; ++i) row[t + i * 256] = f2bf(vals[i] * r);
}

// ---------------------------------------------------------------------------
// K5: ctxT[e,d] = sum_n v[e,n] * k[d,n]  per (b,h).  Both operands naturally
// contiguous in K (=n): fragments straight from global (L2-resident panels).
// ---------------------------------------------------------------------------
__global__ __launch_bounds__(256)
void k_context(const __bf16* __restrict__ qkv, __bf16* __restrict__ ctxT) {
  int bh = blockIdx.x;
  int b = bh >> 3, h = bh & 7;
  int w = threadIdx.x >> 5, L = threadIdx.x & 31, lm = L & 15, lh = L >> 4;
  int ei = w & 3;
  int d0 = (w >> 2) * 32;
  const __bf16* vbase = qkv + ((size_t)(2 * B_ * C_) + (size_t)b * C_ + h * DH) * NPOS;
  const __bf16* kbase = qkv + ((size_t)(1 * B_ * C_) + (size_t)b * C_ + h * DH) * NPOS;
  fx8 acc0 = fx8_zero(), acc1 = fx8_zero();
  for (int n0 = 0; n0 < NPOS; n0 += 32) {
    const __bf16* pa  = vbase + (size_t)(ei * 16 + lm) * NPOS + n0 + lh * 8;
    bfx16 a = load_frag(pa, pa + 16);
    const __bf16* pb0 = kbase + (size_t)(d0 + lm) * NPOS + n0 + lh * 8;
    bfx16 b0 = load_frag(pb0, pb0 + 16);
    const __bf16* pb1 = kbase + (size_t)(d0 + 16 + lm) * NPOS + n0 + lh * 8;
    bfx16 b1 = load_frag(pb1, pb1 + 16);
    acc0 = wmma_bf16(a, b0, acc0);
    acc1 = wmma_bf16(a, b1, acc1);
  }
  __bf16* out = ctxT + (size_t)bh * (DH * DH);
#pragma unroll
  for (int i = 0; i < 8; ++i) {
    int e = ei * 16 + lh * 8 + i;
    out[e * DH + d0 + lm]      = f2bf(acc0[i]);
    out[e * DH + d0 + 16 + lm] = f2bf(acc1[i]);
  }
}

// ---------------------------------------------------------------------------
// K6: attnT[p, h*64+e] = sum_d qT[p, h*64+d] * ctxT[e, d]  per (b,h).
// ---------------------------------------------------------------------------
__global__ __launch_bounds__(256)
void k_apply(const __bf16* __restrict__ qT, const __bf16* __restrict__ ctxT,
             __bf16* __restrict__ attnT) {
  int b = blockIdx.z, h = blockIdx.y, pc = blockIdx.x;
  int w = threadIdx.x >> 5, L = threadIdx.x & 31, lm = L & 15, lh = L >> 4;
  int p0 = pc * 512 + w * 64;
  const __bf16* qb = qT + (size_t)b * NPOS * C_ + h * DH;
  const __bf16* cb = ctxT + (size_t)(b * 8 + h) * (DH * DH);

  fx8 acc[4][4];
#pragma unroll
  for (int pi = 0; pi < 4; ++pi)
#pragma unroll
    for (int ei = 0; ei < 4; ++ei) acc[pi][ei] = fx8_zero();

#pragma unroll
  for (int k0 = 0; k0 < DH; k0 += 32) {
    bfx16 af[4], bfr[4];
#pragma unroll
    for (int pi = 0; pi < 4; ++pi) {
      const __bf16* pa = qb + (size_t)(p0 + pi * 16 + lm) * C_ + k0 + lh * 8;
      af[pi] = load_frag(pa, pa + 16);
    }
#pragma unroll
    for (int ei = 0; ei < 4; ++ei) {
      const __bf16* pb = cb + (ei * 16 + lm) * DH + k0 + lh * 8;
      bfr[ei] = load_frag(pb, pb + 16);
    }
#pragma unroll
    for (int pi = 0; pi < 4; ++pi)
#pragma unroll
      for (int ei = 0; ei < 4; ++ei)
        acc[pi][ei] = wmma_bf16(af[pi], bfr[ei], acc[pi][ei]);
  }

  __bf16* ob = attnT + (size_t)b * NPOS * C_ + h * DH;
#pragma unroll
  for (int pi = 0; pi < 4; ++pi)
#pragma unroll
    for (int ei = 0; ei < 4; ++ei)
#pragma unroll
      for (int i = 0; i < 8; ++i) {
        int p = p0 + pi * 16 + lh * 8 + i;
        int e = ei * 16 + lm;
        ob[(size_t)p * C_ + e] = f2bf(acc[pi][ei][i]);
      }
}

// ---------------------------------------------------------------------------
// K8: final RMS-norm: out = (proj + b_out) * rsqrt(sum^2+eps) * g2 * sqrt(C)
// ---------------------------------------------------------------------------
__global__ void k_final(const float* __restrict__ proj, const float* __restrict__ b_out,
                        const float* __restrict__ g2, float* __restrict__ out) {
  int b = blockIdx.y;
  int p = blockIdx.x * 256 + threadIdx.x;
  const float* pp = proj + (size_t)b * C_ * NPOS + p;
  float ss = 0.f;
  for (int c = 0; c < C_; ++c) { float v = pp[(size_t)c * NPOS] + b_out[c]; ss += v * v; }
  float inv = rsqrtf(ss + 1e-12f) * SQRT_C;
  float* op = out + (size_t)b * C_ * NPOS + p;
  for (int c = 0; c < C_; ++c) {
    float v = pp[(size_t)c * NPOS] + b_out[c];
    op[(size_t)c * NPOS] = v * inv * g2[c];
  }
}

// ---------------------------------------------------------------------------
extern "C" void kernel_launch(void* const* d_in, const int* in_sizes, int n_in,
                              void* d_out, int out_size, void* d_ws, size_t ws_size,
                              hipStream_t stream) {
  const float* x     = (const float*)d_in[0];
  const float* g1    = (const float*)d_in[1];
  const float* w_qkv = (const float*)d_in[2];
  const float* w_out = (const float*)d_in[3];
  const float* b_out = (const float*)d_in[4];
  const float* g2    = (const float*)d_in[5];
  float* out = (float*)d_out;
  char* ws = (char*)d_ws;

  const size_t off_wq   = 0;                              // 1.5 MB
  const size_t off_wo   = off_wq  + (size_t)HID * C_ * 2; // 0.5 MB
  const size_t off_xnT  = off_wo  + (size_t)C_ * C_ * 2;  // 32 MB (reused as qT)
  const size_t off_qkv  = off_xnT + (size_t)B_ * NPOS * C_ * 2; // 96 MB
  const size_t off_ctxT = off_qkv + (size_t)3 * B_ * C_ * NPOS * 2; // 512 KB

  __bf16* wq    = (__bf16*)(ws + off_wq);
  __bf16* wo    = (__bf16*)(ws + off_wo);
  __bf16* xnT   = (__bf16*)(ws + off_xnT);
  __bf16* qT    = (__bf16*)(ws + off_xnT);                 // reuse: xnT dead after GEMM1
  __bf16* qkv   = (__bf16*)(ws + off_qkv);
  __bf16* ctxT  = (__bf16*)(ws + off_ctxT);
  __bf16* attnT = (__bf16*)(ws + off_qkv);                 // reuse: q slab dead after K3
  float*  proj  = (float*)(ws + off_qkv + (size_t)B_ * C_ * NPOS * 2); // k+v slabs

  k_cvt_weights<<<dim3((HID * C_ + C_ * C_) / 256), 256, 0, stream>>>(w_qkv, g1, w_out, wq, wo);
  k_norm_x<<<dim3(NPOS / 256, B_), 256, 0, stream>>>(x, xnT);
  k_gemm<0><<<dim3(NPOS / 256, HID / 128, B_), 256, 0, stream>>>(wq, xnT, qkv);
  k_softmax_q<<<dim3(NPOS / 256, 8, B_), 256, 0, stream>>>(qkv, qT);
  k_softmax_k<<<dim3(C_, B_), 256, 0, stream>>>(qkv);
  k_context<<<dim3(B_ * 8), 256, 0, stream>>>(qkv, ctxT);
  k_apply<<<dim3(8, 8, B_), 256, 0, stream>>>(qT, ctxT, attnT);
  k_gemm<1><<<dim3(NPOS / 256, C_ / 128, B_), 256, 0, stream>>>(wo, attnT, proj);
  k_final<<<dim3(NPOS / 256, B_), 256, 0, stream>>>(proj, b_out, g2, out);
}
